// O3AttentionLayer_1700807049398
// MI455X (gfx1250) — matrix operand order repaired and checked by reading, not compiled, not run on hardware
//
#include <hip/hip_runtime.h>
#include <hip/hip_bf16.h>
#include <math.h>

typedef __attribute__((ext_vector_type(16))) _Float16 v16h;
typedef __attribute__((ext_vector_type(8)))  float    v8f;

#define WAVES_PER_BLOCK 4
#define SQ3f     1.7320508075688772f
#define INV_SQ3f 0.5773502691896258f

// Per-wave LDS working set: ~10.2 KB  (x4 waves = ~41 KB)
struct WaveLDS {
  _Float16 hk[16][16];   // silu(rad@Wk1) staged for GEMM2-A transpose
  _Float16 hv[16][16];   // silu(rad@Wv1)
  float xs[16][16];      // per-edge scalar features of src
  float xv[16][24];      // per-edge vector features of src (8x3)
  float y1[16][3];       // SQ3 * unit edge vector
  float a24[16][24];     // scalar-path coefficients: [xs ; (xv.y1)/sqrt3]
  float qts[16][8];      // q_s[dst] @ Ws_ss
  float qtv[16][12];     // (q_v[dst] @ Ws_vv) [j*3+c]
  float accKs[16][8];    // k_s accumulator (pre inv)
  float accKv[16][12];   // k_v accumulator [o*3+c]
  float accVs[16][16];   // v_s accumulator
  float accVv[16][24];   // v_v accumulator [o*3+c]
};

__global__ void o3_zero_kernel(float* __restrict__ out, float* __restrict__ Z, int N) {
  int idx = blockIdx.x * blockDim.x + threadIdx.x;
  int n40 = N * 40;
  if (idx < n40) out[idx] = 0.f;
  else if (idx < n40 + N) Z[idx - n40] = 0.f;
}

__global__ void o3_norm_kernel(float* __restrict__ out, const float* __restrict__ Z, int N) {
  int idx = blockIdx.x * blockDim.x + threadIdx.x;
  if (idx < N * 40) {
    float z = Z[idx / 40];
    if (z > 0.f) out[idx] /= z;
  }
}

__global__ __launch_bounds__(128) void o3_edge_kernel(
    const float* __restrict__ pos,  const float* __restrict__ x,
    const float* __restrict__ Wq_s, const float* __restrict__ Wq_v,
    const float* __restrict__ Wk1,  const float* __restrict__ Wk2,
    const float* __restrict__ Wv1,  const float* __restrict__ Wv2,
    const float* __restrict__ Ws_ss,const float* __restrict__ Ws_vv,
    const int* __restrict__ esrc,   const int* __restrict__ edst,
    float* __restrict__ out, float* __restrict__ Z, int E)
{
  __shared__ WaveLDS smem[WAVES_PER_BLOCK];
  const int wave  = threadIdx.x >> 5;
  const int lane  = threadIdx.x & 31;
  const int row   = lane & 15;      // edge row within tile / N column for B,C,D
  const int khalf = lane >> 4;      // which K half this lane carries
  WaveLDS& W = smem[wave];

  const int tileBase = (blockIdx.x * WAVES_PER_BLOCK + wave) * 16;
  const int eIdx  = tileBase + row;
  const bool valid = (eIdx < E);
  const int  ec    = valid ? eIdx : 0;

  const int src = esrc[ec];
  const int dst = edst[ec];

  // ---- edge geometry (both lane-halves compute their edge's radials) ----
  float vx = pos[dst*3+0] - pos[src*3+0];
  float vy = pos[dst*3+1] - pos[src*3+1];
  float vz = pos[dst*3+2] - pos[src*3+2];
  float d  = sqrtf(vx*vx + vy*vy + vz*vz);
  float dsafe = fmaxf(d, 1e-6f);
  float rinv  = __builtin_amdgcn_rcpf(dsafe);           // single v_rcp_f32
  float y1c0 = SQ3f * vx * rinv, y1c1 = SQ3f * vy * rinv, y1c2 = SQ3f * vz * rinv;
  // rad[k] = sqrt(2/R)*sqrt(32)*sin((k+1)*pi*d/R)/d  (0 beyond cutoff)
  float rscale = (d < 2.5f) ? (5.059644256269407f * rinv) : 0.f;  // sqrt(2/2.5)*sqrt(32)
  float arg = 3.14159265358979323846f * dsafe * 0.4f;   // pi*d/R

  // A operand (16 edges x 32 basis, f16). contiguous k order, same as B:
  // lanes 0-15: K=0..15 (VGPR v: K=2v,2v+1); lanes 16-31: K=16..31
  v16h aR;
  #pragma unroll
  for (int v = 0; v < 8; ++v) {
    int k0 = 2*v + 16*khalf;
    aR[2*v]   = (_Float16)(rscale * __sinf((float)(k0+1) * arg));
    aR[2*v+1] = (_Float16)(rscale * __sinf((float)(k0+2) * arg));
  }

  // ---- per-edge setup (one lane per edge) ----
  if (khalf == 0) {
    const float* xr = x + (size_t)src * 40;
    #pragma unroll
    for (int i = 0; i < 16; ++i) { float s = xr[i]; W.xs[row][i] = s; W.a24[row][i] = s; }
    #pragma unroll
    for (int p = 0; p < 8; ++p) {
      float v0 = xr[16+p*3+0], v1 = xr[16+p*3+1], v2 = xr[16+p*3+2];
      W.xv[row][p*3+0] = v0; W.xv[row][p*3+1] = v1; W.xv[row][p*3+2] = v2;
      W.a24[row][16+p] = (v0*y1c0 + v1*y1c1 + v2*y1c2) * INV_SQ3f;
    }
    W.y1[row][0] = y1c0; W.y1[row][1] = y1c1; W.y1[row][2] = y1c2;

    // q at destination, pre-contracted with similarity matrices
    const float* xd = x + (size_t)dst * 40;
    float qs[8];
    #pragma unroll
    for (int i = 0; i < 8; ++i) {
      float s = 0.f;
      #pragma unroll
      for (int p = 0; p < 16; ++p) s += xd[p] * Wq_s[p*8+i];
      qs[i] = s * 0.25f;                       // 1/sqrt(16)
    }
    #pragma unroll
    for (int j = 0; j < 8; ++j) {
      float s = 0.f;
      #pragma unroll
      for (int i = 0; i < 8; ++i) s += qs[i] * Ws_ss[i*8+j];
      W.qts[row][j] = s;
    }
    float qv[12];
    #pragma unroll
    for (int i = 0; i < 4; ++i)
      #pragma unroll
      for (int c = 0; c < 3; ++c) {
        float s = 0.f;
        #pragma unroll
        for (int p = 0; p < 8; ++p) s += xd[16+p*3+c] * Wq_v[p*4+i];
        qv[i*3+c] = s * 0.35355339059327373f;  // 1/sqrt(8)
      }
    #pragma unroll
    for (int j = 0; j < 4; ++j)
      #pragma unroll
      for (int c = 0; c < 3; ++c) {
        float s = 0.f;
        #pragma unroll
        for (int i = 0; i < 4; ++i) s += qv[i*3+c] * Ws_vv[i*4+j];
        W.qtv[row][j*3+c] = s;
      }
    #pragma unroll
    for (int o = 0; o < 8;  ++o) W.accKs[row][o] = 0.f;
    #pragma unroll
    for (int o = 0; o < 12; ++o) W.accKv[row][o] = 0.f;
    #pragma unroll
    for (int o = 0; o < 16; ++o) W.accVs[row][o] = 0.f;
    #pragma unroll
    for (int o = 0; o < 24; ++o) W.accVv[row][o] = 0.f;
  }
  __syncthreads();

  // ---- GEMM1: rad(16x32) @ W1(32x16) for both k and v nets ----
  // B layout: lane holds N=lane%16; lanes 0-15 K=0..15 (VGPR v: K=2v,2v+1), lanes 16-31 K=16..31
  const float s1 = 0.17677669529663687f;       // 1/sqrt(32)
  v16h b1;
  #pragma unroll
  for (int v = 0; v < 8; ++v) {
    int k0 = 2*v + 16*khalf;
    b1[2*v]   = (_Float16)(Wk1[k0*16 + row] * s1);
    b1[2*v+1] = (_Float16)(Wk1[(k0+1)*16 + row] * s1);
  }
  v8f ck = {0.f,0.f,0.f,0.f,0.f,0.f,0.f,0.f};
  ck = __builtin_amdgcn_wmma_f32_16x16x32_f16(false, aR, false, b1, (short)0, ck, false, false);
  #pragma unroll
  for (int v = 0; v < 8; ++v) {
    int k0 = 2*v + 16*khalf;
    b1[2*v]   = (_Float16)(Wv1[k0*16 + row] * s1);
    b1[2*v+1] = (_Float16)(Wv1[(k0+1)*16 + row] * s1);
  }
  v8f cv = {0.f,0.f,0.f,0.f,0.f,0.f,0.f,0.f};
  cv = __builtin_amdgcn_wmma_f32_16x16x32_f16(false, aR, false, b1, (short)0, cv, false, false);

  // silu (x * rcp(1+exp(-x)) -> v_exp + v_rcp, no IEEE divide), stage h in LDS
  // D layout: lane holds N=lane%16, M=r+8*khalf
  #pragma unroll
  for (int r = 0; r < 8; ++r) {
    int er = r + 8*khalf;
    float h0 = ck[r]; h0 = h0 * __builtin_amdgcn_rcpf(1.f + __expf(-h0));
    float h1 = cv[r]; h1 = h1 * __builtin_amdgcn_rcpf(1.f + __expf(-h1));
    W.hk[er][row] = (_Float16)h0;
    W.hv[er][row] = (_Float16)h1;
  }
  __syncthreads();

  // A operands for GEMM2 (K=16 zero-padded to 32)
  v16h a2k, a2v;
  #pragma unroll
  for (int v = 0; v < 8; ++v) {
    if (khalf == 0) {
      int k0 = 2*v;
      a2k[2*v] = W.hk[row][k0]; a2k[2*v+1] = W.hk[row][k0+1];
      a2v[2*v] = W.hv[row][k0]; a2v[2*v+1] = W.hv[row][k0+1];
    } else {
      a2k[2*v] = (_Float16)0.f; a2k[2*v+1] = (_Float16)0.f;
      a2v[2*v] = (_Float16)0.f; a2v[2*v+1] = (_Float16)0.f;
    }
  }

  const float s2 = 0.25f;                      // 1/sqrt(16)

  // ---- GEMM2 + equivariant contraction: wk (288 cols, 18 tiles) ----
  for (int t = 0; t < 18; ++t) {
    int cn = t*16 + row;
    v16h b2;
    #pragma unroll
    for (int v = 0; v < 8; ++v) {
      float w0 = khalf ? 0.f : Wk2[(2*v)  *288 + cn] * s2;
      float w1 = khalf ? 0.f : Wk2[(2*v+1)*288 + cn] * s2;
      b2[2*v] = (_Float16)w0; b2[2*v+1] = (_Float16)w1;
    }
    v8f dk = {0.f,0.f,0.f,0.f,0.f,0.f,0.f,0.f};
    dk = __builtin_amdgcn_wmma_f32_16x16x32_f16(false, a2k, false, b2, (short)0, dk, false, false);
    #pragma unroll
    for (int r = 0; r < 8; ++r) {
      int er = r + 8*khalf;
      float val = dk[r];
      int n = cn;
      if (n < 192) {                 // w_ss(16x8) ++ w_vv(8x8): scalar output path
        int i = n >> 3, o = n & 7;
        atomicAdd(&W.accKs[er][o], W.a24[er][i] * val);
      } else {                       // w_sv(16x4) ++ w_vs(8x4): vector output path
        int m = n - 192; int i = m >> 2, o = m & 3;
        #pragma unroll
        for (int c = 0; c < 3; ++c) {
          float bcf = (i < 16) ? (W.xs[er][i] * W.y1[er][c]) : W.xv[er][(i-16)*3 + c];
          atomicAdd(&W.accKv[er][o*3+c], bcf * val);
        }
      }
    }
  }

  // ---- GEMM2 + contraction: wv (576 cols, 36 tiles) ----
  for (int t = 0; t < 36; ++t) {
    int cn = t*16 + row;
    v16h b2;
    #pragma unroll
    for (int v = 0; v < 8; ++v) {
      float w0 = khalf ? 0.f : Wv2[(2*v)  *576 + cn] * s2;
      float w1 = khalf ? 0.f : Wv2[(2*v+1)*576 + cn] * s2;
      b2[2*v] = (_Float16)w0; b2[2*v+1] = (_Float16)w1;
    }
    v8f dv = {0.f,0.f,0.f,0.f,0.f,0.f,0.f,0.f};
    dv = __builtin_amdgcn_wmma_f32_16x16x32_f16(false, a2v, false, b2, (short)0, dv, false, false);
    #pragma unroll
    for (int r = 0; r < 8; ++r) {
      int er = r + 8*khalf;
      float val = dv[r];
      int n = cn;
      if (n < 384) {                 // w_ss(16x16) ++ w_vv(8x16)
        int i = n >> 4, o = n & 15;
        atomicAdd(&W.accVs[er][o], W.a24[er][i] * val);
      } else {                       // w_sv(16x8) ++ w_vs(8x8)
        int m = n - 384; int i = m >> 3, o = m & 7;
        #pragma unroll
        for (int c = 0; c < 3; ++c) {
          float bcf = (i < 16) ? (W.xs[er][i] * W.y1[er][c]) : W.xv[er][(i-16)*3 + c];
          atomicAdd(&W.accVv[er][o*3+c], bcf * val);
        }
      }
    }
  }
  __syncthreads();

  // ---- finalize: similarity, exp, scatter attn-weighted values ----
  if (khalf == 0 && valid) {
    const float inv = 0.2041241452319315f;     // 1/sqrt(16+8)
    float sim = 0.f;
    #pragma unroll
    for (int o = 0; o < 8; ++o) sim += W.qts[row][o] * (W.accKs[row][o] * inv);
    float simv = 0.f;
    #pragma unroll
    for (int j = 0; j < 12; ++j) simv += W.qtv[row][j] * (W.accKv[row][j] * inv);
    sim = (sim + simv * INV_SQ3f) * 0.11180339887498948f;  // 1/sqrt(8*8+4*4)
    float attn = __expf(sim);
    atomicAdd(&Z[dst], attn);
    float* od = out + (size_t)dst * 40;
    #pragma unroll
    for (int o = 0; o < 16; ++o) atomicAdd(&od[o],      attn * W.accVs[row][o] * inv);
    #pragma unroll
    for (int j = 0; j < 24; ++j) atomicAdd(&od[16 + j], attn * W.accVv[row][j] * inv);
  }
}

extern "C" void kernel_launch(void* const* d_in, const int* in_sizes, int n_in,
                              void* d_out, int out_size, void* d_ws, size_t ws_size,
                              hipStream_t stream) {
  const float* pos   = (const float*)d_in[0];
  const float* x     = (const float*)d_in[1];
  const float* Wq_s  = (const float*)d_in[2];
  const float* Wq_v  = (const float*)d_in[3];
  const float* Wk1   = (const float*)d_in[4];
  const float* Wk2   = (const float*)d_in[5];
  const float* Wv1   = (const float*)d_in[6];
  const float* Wv2   = (const float*)d_in[7];
  const float* Ws_ss = (const float*)d_in[8];
  const float* Ws_vv = (const float*)d_in[9];
  const int*   esrc  = (const int*)d_in[10];
  const int*   edst  = (const int*)d_in[11];

  const int N = in_sizes[0] / 3;
  const int E = in_sizes[10];

  float* out = (float*)d_out;
  float* Z   = (float*)d_ws;            // N floats of scratch for softmax denominator

  int zthreads = N * 41;
  o3_zero_kernel<<<(zthreads + 255) / 256, 256, 0, stream>>>(out, Z, N);

  int blocks = (E + WAVES_PER_BLOCK * 16 - 1) / (WAVES_PER_BLOCK * 16);
  o3_edge_kernel<<<blocks, 128, 0, stream>>>(pos, x, Wq_s, Wq_v, Wk1, Wk2, Wv1, Wv2,
                                             Ws_ss, Ws_vv, esrc, edst, out, Z, E);

  o3_norm_kernel<<<(N * 40 + 255) / 256, 256, 0, stream>>>(out, Z, N);
}
